// Discriminator_10694468567287
// MI455X (gfx1250) — matrix operand order repaired and checked by reading, not compiled, not run on hardware
//
#include <hip/hip_runtime.h>

#define B_   16
#define L_   512
#define H_   1024
#define NH_  16
#define DK_  64
#define NL_  6
#define O_   4
#define M_   (B_ * L_)          /* 8192 rows */
#define EPS_ 1e-5f

typedef __attribute__((ext_vector_type(16))) __bf16 bf16x16;
typedef __attribute__((ext_vector_type(8)))  float  f32x8;

union FragU {
    bf16x16        v;
    uint4          q[2];
    unsigned short h[16];
};

__device__ __forceinline__ unsigned short f32_to_bf16(float f) {
    unsigned int u = __float_as_uint(f);
    u += 0x7FFFu + ((u >> 16) & 1u);          // round-to-nearest-even
    return (unsigned short)(u >> 16);
}

// A-matrix fragment (16x32 bf16, ISA 7.12.2): lane m=idx; regs hold K = kb0..kb0+7
// and K = 16+kb0..16+kb0+7 where kb0 = 8*half.
__device__ __forceinline__ bf16x16 load16_split(const unsigned short* p) {
    FragU f;
    f.q[0] = *(const uint4*)(p);
    f.q[1] = *(const uint4*)(p + 16);
    return f.v;
}
// A/B fragment (contiguous k, same for A and B): lane m/n=idx; regs hold K = 16*half .. 16*half+15.
__device__ __forceinline__ bf16x16 load16_contig(const unsigned short* p) {
    FragU f;
    f.q[0] = *(const uint4*)(p);
    f.q[1] = *(const uint4*)(p + 8);
    return f.v;
}

__device__ __forceinline__ bf16x16 lds_a_frag(const unsigned short* base, int r0, int lane) {
    const int half = lane >> 4, idx = lane & 15;
    return load16_contig(base + (r0 + idx) * 32 + half * 16);
}
__device__ __forceinline__ bf16x16 lds_b_frag(const unsigned short* base, int c0, int lane) {
    const int half = lane >> 4, idx = lane & 15;
    return load16_contig(base + (c0 + idx) * 32 + half * 16);
}

// ---------------------------------------------------------------------------
// CDNA5 async global->LDS copy (ASYNCcnt tracked), 16B per lane per issue.
// LDS operand = low 32 bits of the flat shared pointer (== LDS byte offset).
// ---------------------------------------------------------------------------
__device__ __forceinline__ void async_ld16(unsigned lds_addr, const unsigned short* g) {
    asm volatile("global_load_async_to_lds_b128 %0, %1, off"
                 :: "v"(lds_addr), "v"((unsigned long long)(size_t)g)
                 : "memory");
}
__device__ __forceinline__ void wait_async_le3() {
    asm volatile("s_wait_asynccnt 0x3" ::: "memory");
}
__device__ __forceinline__ void wait_async_0() {
    asm volatile("s_wait_asynccnt 0x0" ::: "memory");
}

// ---------------------------------------------------------------------------
// Weight conversion f32 -> bf16 (once per launch; removes conversion from GEMM)
// ---------------------------------------------------------------------------
__global__ __launch_bounds__(256) void cvt_bf16_kernel(const float* __restrict__ in,
                                                       unsigned short* __restrict__ out,
                                                       int n4) {   // n/4 groups
    const int g = blockIdx.x * 256 + threadIdx.x;
    if (g < n4) {
        const float4 f = *(const float4*)(in + (size_t)g * 4);
        uint2 u;
        u.x = (unsigned)f32_to_bf16(f.x) | ((unsigned)f32_to_bf16(f.y) << 16);
        u.y = (unsigned)f32_to_bf16(f.z) | ((unsigned)f32_to_bf16(f.w) << 16);
        *(uint2*)(out + (size_t)g * 4) = u;
    }
}

// ---------------------------------------------------------------------------
// Embedding + sinusoidal positional encoding
// ---------------------------------------------------------------------------
__global__ __launch_bounds__(256) void embed_kernel(const int* __restrict__ x,
                                                    const float* __restrict__ emb,
                                                    float* __restrict__ z) {
    const int bl  = blockIdx.x;
    const int l   = bl % L_;
    const int tok = x[bl];
    for (int i = threadIdx.x; i < H_; i += 256) {
        const int   i2  = i >> 1;
        const float div = __powf(10000.f, (float)i2 / (float)H_);
        const float ang = (float)l / div;
        const float pe  = (i & 1) ? __cosf(ang) : __sinf(ang);
        z[(size_t)bl * H_ + i] = emb[(size_t)tok * H_ + i] + pe;
    }
}

// ---------------------------------------------------------------------------
// LayerNorm (f32 in) -> bf16 out
// ---------------------------------------------------------------------------
__global__ __launch_bounds__(256) void layernorm_kernel(const float* __restrict__ z,
                                                        const float* __restrict__ w,
                                                        const float* __restrict__ bvec,
                                                        unsigned short* __restrict__ out) {
    __shared__ float red[256];
    __shared__ float stats[2];
    const int row = blockIdx.x, tid = threadIdx.x;
    const float* zr = z + (size_t)row * H_;
    float s = 0.f, s2 = 0.f;
    for (int i = tid; i < H_; i += 256) { float v = zr[i]; s += v; s2 += v * v; }
    red[tid] = s; __syncthreads();
    for (int off = 128; off > 0; off >>= 1) { if (tid < off) red[tid] += red[tid + off]; __syncthreads(); }
    if (tid == 0) stats[0] = red[0] / (float)H_;
    __syncthreads();
    red[tid] = s2; __syncthreads();
    for (int off = 128; off > 0; off >>= 1) { if (tid < off) red[tid] += red[tid + off]; __syncthreads(); }
    if (tid == 0) {
        const float mean = stats[0];
        stats[1] = rsqrtf(red[0] / (float)H_ - mean * mean + EPS_);
    }
    __syncthreads();
    const float mean = stats[0], rstd = stats[1];
    unsigned short* orow = out + (size_t)row * H_;
    for (int i = tid; i < H_; i += 256)
        orow[i] = f32_to_bf16((zr[i] - mean) * rstd * w[i] + bvec[i]);
}

// ---------------------------------------------------------------------------
// GEMM: C[M,N] = A[M,K](bf16) * Wb[N,K](bf16)^T  (+bias/relu/residual)
// 256 threads = 8 waves (4x2); block tile 128x64, K-step 32; wave -> 32x32
// (4 WMMAs, 2x2 fragment reuse).  Double-buffered LDS fed by
// global_load_async_to_lds_b128, pipelined with s_wait_asynccnt.
// ---------------------------------------------------------------------------
template <bool RELU, bool RES, bool OUT_BF16>
__global__ __launch_bounds__(256) void gemm_bf16_kernel(const unsigned short* __restrict__ A,
                                                        const unsigned short* __restrict__ Wb,
                                                        const float* __restrict__ bias,
                                                        const float* __restrict__ res,
                                                        float* __restrict__ outF,
                                                        unsigned short* __restrict__ outB,
                                                        int N, int K) {
    __shared__ __align__(16) unsigned short As[2][128][32];
    __shared__ __align__(16) unsigned short Bs[2][64][32];
    const int tid  = threadIdx.x;
    const int wave = tid >> 5, lane = tid & 31;
    const int half = lane >> 4, idx = lane & 15;
    const int n0 = blockIdx.x * 64, m0 = blockIdx.y * 128;
    const int mw = (wave & 3) * 32;        // wave row offset: 4 waves cover 128 rows
    const int nw = (wave >> 2) * 32;       // wave col offset: 2 waves cover 64 cols

    // staging map: thread -> one 16B chunk of each of: A rows [0,64), A rows [64,128), B rows [0,64)
    const int lr = tid >> 2, lc = (tid & 3) * 8;
    const unsigned short* gA0 = A  + (size_t)(m0 + lr) * K + lc;
    const unsigned short* gA1 = A  + (size_t)(m0 + 64 + lr) * K + lc;
    const unsigned short* gB  = Wb + (size_t)(n0 + lr) * K + lc;
    unsigned ldsA0[2], ldsA1[2], ldsB[2];
#pragma unroll
    for (int b = 0; b < 2; ++b) {
        ldsA0[b] = (unsigned)(size_t)&As[b][lr][lc];
        ldsA1[b] = (unsigned)(size_t)&As[b][64 + lr][lc];
        ldsB[b]  = (unsigned)(size_t)&Bs[b][lr][lc];
    }

    f32x8 acc[2][2];
#pragma unroll
    for (int tm = 0; tm < 2; ++tm)
#pragma unroll
        for (int tn = 0; tn < 2; ++tn)
#pragma unroll
            for (int r = 0; r < 8; ++r) acc[tm][tn][r] = 0.f;

    // prologue: stage tile 0
    async_ld16(ldsA0[0], gA0);
    async_ld16(ldsA1[0], gA1);
    async_ld16(ldsB[0],  gB);

    int buf = 0;
    for (int k0 = 0; k0 < K; k0 += 32) {
        if (k0 + 32 < K) {   // stage next tile into the other buffer, then wait for current
            async_ld16(ldsA0[buf ^ 1], gA0 + k0 + 32);
            async_ld16(ldsA1[buf ^ 1], gA1 + k0 + 32);
            async_ld16(ldsB[buf ^ 1],  gB  + k0 + 32);
            wait_async_le3();
        } else {
            wait_async_0();
        }
        __syncthreads();
        const unsigned short* as = &As[buf][0][0];
        const unsigned short* bs = &Bs[buf][0][0];
        const bf16x16 a0 = lds_a_frag(as, mw, lane);
        const bf16x16 a1 = lds_a_frag(as, mw + 16, lane);
        const bf16x16 b0 = lds_b_frag(bs, nw, lane);
        const bf16x16 b1 = lds_b_frag(bs, nw + 16, lane);
        acc[0][0] = __builtin_amdgcn_wmma_f32_16x16x32_bf16(false, a0, false, b0, (short)0, acc[0][0], false, false);
        acc[0][1] = __builtin_amdgcn_wmma_f32_16x16x32_bf16(false, a0, false, b1, (short)0, acc[0][1], false, false);
        acc[1][0] = __builtin_amdgcn_wmma_f32_16x16x32_bf16(false, a1, false, b0, (short)0, acc[1][0], false, false);
        acc[1][1] = __builtin_amdgcn_wmma_f32_16x16x32_bf16(false, a1, false, b1, (short)0, acc[1][1], false, false);
        __syncthreads();   // all waves done reading buf before it is overwritten next iter
        buf ^= 1;
    }

#pragma unroll
    for (int tm = 0; tm < 2; ++tm)
#pragma unroll
        for (int tn = 0; tn < 2; ++tn) {
            const int   gcol = n0 + nw + tn * 16 + idx;
            const float bv   = bias ? bias[gcol] : 0.f;
#pragma unroll
            for (int r = 0; r < 8; ++r) {
                float val = acc[tm][tn][r] + bv;
                const size_t grow = (size_t)(m0 + mw + tm * 16 + r + half * 8);
                if (RELU) val = fmaxf(val, 0.f);
                if (RES)  val += res[grow * N + gcol];
                if (OUT_BF16) outB[grow * N + gcol] = f32_to_bf16(val);
                else          outF[grow * N + gcol] = val;
            }
        }
}

// ---------------------------------------------------------------------------
// Attention: one wave per (b, head, 16-query tile); flash-style online softmax.
// ---------------------------------------------------------------------------
__global__ __launch_bounds__(32) void attn_kernel(const unsigned short* __restrict__ q,
                                                  const unsigned short* __restrict__ k,
                                                  const unsigned short* __restrict__ v,
                                                  const int* __restrict__ length,
                                                  unsigned short* __restrict__ c) {
    __shared__ __align__(16) unsigned short Ps[16][32];
    const int lane = threadIdx.x;
    const int half = lane >> 4, idx = lane & 15;
    const int qt = blockIdx.x, h = blockIdx.y, b = blockIdx.z;
    const int len   = length[b];
    const int hbase = h * DK_;
    const size_t rowQ = (size_t)b * L_ + qt * 16;
    const float  scale = 0.125f;  // 1/sqrt(64)

    const bf16x16 qf0 = load16_contig(q + (rowQ + idx) * H_ + hbase + 0  + half * 16);
    const bf16x16 qf1 = load16_contig(q + (rowQ + idx) * H_ + hbase + 32 + half * 16);

    float mrow[8], lrow[8];
    f32x8 oacc[4];
#pragma unroll
    for (int r = 0; r < 8; ++r) {
        mrow[r] = -1e30f; lrow[r] = 0.f;
        oacc[0][r] = 0.f; oacc[1][r] = 0.f; oacc[2][r] = 0.f; oacc[3][r] = 0.f;
    }

    for (int kb = 0; kb < L_; kb += 32) {
        f32x8 s0, s1;
#pragma unroll
        for (int r = 0; r < 8; ++r) { s0[r] = 0.f; s1[r] = 0.f; }
        {   // S tiles: keys kb..kb+15 and kb+16..kb+31, K-dim = DK=64 (2 WMMA steps each)
            const size_t kr0 = ((size_t)b * L_ + kb + idx) * H_ + hbase;
            const bf16x16 kf0 = load16_contig(k + kr0 + 0  + half * 16);
            const bf16x16 kf1 = load16_contig(k + kr0 + 32 + half * 16);
            s0 = __builtin_amdgcn_wmma_f32_16x16x32_bf16(false, qf0, false, kf0, (short)0, s0, false, false);
            s0 = __builtin_amdgcn_wmma_f32_16x16x32_bf16(false, qf1, false, kf1, (short)0, s0, false, false);
            const size_t kr1 = ((size_t)b * L_ + kb + 16 + idx) * H_ + hbase;
            const bf16x16 kg0 = load16_contig(k + kr1 + 0  + half * 16);
            const bf16x16 kg1 = load16_contig(k + kr1 + 32 + half * 16);
            s1 = __builtin_amdgcn_wmma_f32_16x16x32_bf16(false, qf0, false, kg0, (short)0, s1, false, false);
            s1 = __builtin_amdgcn_wmma_f32_16x16x32_bf16(false, qf1, false, kg1, (short)0, s1, false, false);
        }
        const bool mask0 = (kb + idx)      >= len;
        const bool mask1 = (kb + 16 + idx) >= len;
#pragma unroll
        for (int r = 0; r < 8; ++r) {
            float a = s0[r] * scale; if (mask0) a = -1e30f;
            float e = s1[r] * scale; if (mask1) e = -1e30f;
            float tm = fmaxf(a, e);
#pragma unroll
            for (int m = 1; m < 16; m <<= 1) tm = fmaxf(tm, __shfl_xor(tm, m, 32));
            const float mnew = fmaxf(mrow[r], tm);
            const float corr = __expf(mrow[r] - mnew);
            mrow[r] = mnew;
            const float p0 = __expf(a - mnew);
            const float p1 = __expf(e - mnew);
            s0[r] = p0; s1[r] = p1;
            float ps = p0 + p1;
#pragma unroll
            for (int m = 1; m < 16; m <<= 1) ps += __shfl_xor(ps, m, 32);
            lrow[r] = lrow[r] * corr + ps;
            oacc[0][r] *= corr; oacc[1][r] *= corr; oacc[2][r] *= corr; oacc[3][r] *= corr;
        }
        // transpose P (C-layout f32 -> A-layout bf16) through LDS
#pragma unroll
        for (int r = 0; r < 8; ++r) {
            Ps[r + half * 8][idx]      = f32_to_bf16(s0[r]);
            Ps[r + half * 8][16 + idx] = f32_to_bf16(s1[r]);
        }
        __syncthreads();
        const bf16x16 pf = load16_contig(&Ps[idx][half * 16]);
        __syncthreads();
        // O += P(16x32) * V(32x64) as four 16x16 n-tiles
#pragma unroll
        for (int t4 = 0; t4 < 4; ++t4) {
            FragU vf;
            const unsigned short* vp =
                v + ((size_t)b * L_ + kb + half * 16) * H_ + hbase + t4 * 16 + idx;
#pragma unroll
            for (int j = 0; j < 16; ++j) vf.h[j] = vp[(size_t)j * H_];
            oacc[t4] = __builtin_amdgcn_wmma_f32_16x16x32_bf16(false, pf, false, vf.v, (short)0, oacc[t4], false, false);
        }
    }

#pragma unroll
    for (int t4 = 0; t4 < 4; ++t4)
#pragma unroll
        for (int r = 0; r < 8; ++r) {
            const float  val = oacc[t4][r] / lrow[r];
            const size_t row = rowQ + r + half * 8;
            c[row * H_ + hbase + t4 * 16 + idx] = f32_to_bf16(val);
        }
}

// ---------------------------------------------------------------------------
// Generator on token 0: LN + 4-way projection + log_softmax
// ---------------------------------------------------------------------------
__global__ __launch_bounds__(256) void final_kernel(const float* __restrict__ z,
                                                    const float* __restrict__ gw,
                                                    const float* __restrict__ gb,
                                                    const float* __restrict__ W,
                                                    const float* __restrict__ bias,
                                                    float* __restrict__ out) {
    __shared__ float red[256];
    __shared__ float stats[2];
    __shared__ float logits[O_];
    const int b = blockIdx.x, tid = threadIdx.x;
    const float* zr = z + (size_t)b * L_ * H_;   // token 0 row of batch b
    float s = 0.f, s2 = 0.f;
    for (int i = tid; i < H_; i += 256) { float v = zr[i]; s += v; s2 += v * v; }
    red[tid] = s; __syncthreads();
    for (int off = 128; off > 0; off >>= 1) { if (tid < off) red[tid] += red[tid + off]; __syncthreads(); }
    if (tid == 0) stats[0] = red[0] / (float)H_;
    __syncthreads();
    red[tid] = s2; __syncthreads();
    for (int off = 128; off > 0; off >>= 1) { if (tid < off) red[tid] += red[tid + off]; __syncthreads(); }
    if (tid == 0) {
        const float mean = stats[0];
        stats[1] = rsqrtf(red[0] / (float)H_ - mean * mean + EPS_);
    }
    __syncthreads();
    const float mean = stats[0], rstd = stats[1];
    for (int o = 0; o < O_; ++o) {
        float d = 0.f;
        for (int i = tid; i < H_; i += 256) {
            const float nv = (zr[i] - mean) * rstd * gw[i] + gb[i];
            d += nv * W[(size_t)o * H_ + i];
        }
        __syncthreads();
        red[tid] = d; __syncthreads();
        for (int off = 128; off > 0; off >>= 1) { if (tid < off) red[tid] += red[tid + off]; __syncthreads(); }
        if (tid == 0) logits[o] = red[0] + bias[o];
        __syncthreads();
    }
    if (tid == 0) {
        float m = logits[0];
        for (int o = 1; o < O_; ++o) m = fmaxf(m, logits[o]);
        float se = 0.f;
        for (int o = 0; o < O_; ++o) se += __expf(logits[o] - m);
        const float lse = m + __logf(se);
        for (int o = 0; o < O_; ++o) out[b * O_ + o] = logits[o] - lse;
    }
}

// ---------------------------------------------------------------------------
extern "C" void kernel_launch(void* const* d_in, const int* in_sizes, int n_in,
                              void* d_out, int out_size, void* d_ws, size_t ws_size,
                              hipStream_t stream) {
    (void)in_sizes; (void)n_in; (void)out_size; (void)ws_size;
    const int*   x      = (const int*)  d_in[0];
    const int*   length = (const int*)  d_in[1];
    const float* emb    = (const float*)d_in[2];
    const float* Wq     = (const float*)d_in[3];
    const float* Wk     = (const float*)d_in[4];
    const float* Wv     = (const float*)d_in[5];
    const float* Wo     = (const float*)d_in[6];
    const float* ln1w   = (const float*)d_in[7];
    const float* ln1b   = (const float*)d_in[8];
    const float* ln2w   = (const float*)d_in[9];
    const float* ln2b   = (const float*)d_in[10];
    const float* fc1w   = (const float*)d_in[11];
    const float* fc1b   = (const float*)d_in[12];
    const float* fc2w   = (const float*)d_in[13];
    const float* fc2b   = (const float*)d_in[14];
    const float* glnw   = (const float*)d_in[15];
    const float* glnb   = (const float*)d_in[16];
    const float* gw     = (const float*)d_in[17];
    const float* gb     = (const float*)d_in[18];
    float* out = (float*)d_out;

    // Workspace layout (~336 MB)
    char* ws = (char*)d_ws;
    const size_t MH  = (size_t)M_ * H_;
    const size_t WHH = (size_t)NL_ * H_ * H_;      // 6*H*H
    float*          z    = (float*)ws;           ws += MH * 4;
    unsigned short* zn   = (unsigned short*)ws;  ws += MH * 2;
    unsigned short* qbuf = (unsigned short*)ws;  ws += MH * 2;
    unsigned short* kbuf = (unsigned short*)ws;  ws += MH * 2;
    unsigned short* vbuf = (unsigned short*)ws;  ws += MH * 2;
    unsigned short* cbuf = (unsigned short*)ws;  ws += MH * 2;
    unsigned short* fbuf = (unsigned short*)ws;  ws += MH * 8;      // [M, 4H] bf16
    unsigned short* Wqb  = (unsigned short*)ws;  ws += WHH * 2;
    unsigned short* Wkb  = (unsigned short*)ws;  ws += WHH * 2;
    unsigned short* Wvb  = (unsigned short*)ws;  ws += WHH * 2;
    unsigned short* Wob  = (unsigned short*)ws;  ws += WHH * 2;
    unsigned short* F1b  = (unsigned short*)ws;  ws += WHH * 8;     // 6*4H*H bf16
    unsigned short* F2b  = (unsigned short*)ws;  ws += WHH * 8;     // 6*H*4H bf16

    const dim3 blk(256);
    const dim3 gH(H_ / 64, M_ / 128);         // N=1024 GEMMs
    const dim3 gF(4 * H_ / 64, M_ / 128);     // N=4096 GEMM
    const dim3 gA(L_ / 16, NH_, B_);          // attention waves

    // one-time (per launch) weight conversion to bf16
    {
        const int nHH4 = (int)(WHH / 4), nF4 = (int)(WHH * 4 / 4);
        cvt_bf16_kernel<<<(nHH4 + 255) / 256, blk, 0, stream>>>(Wq,   Wqb, nHH4);
        cvt_bf16_kernel<<<(nHH4 + 255) / 256, blk, 0, stream>>>(Wk,   Wkb, nHH4);
        cvt_bf16_kernel<<<(nHH4 + 255) / 256, blk, 0, stream>>>(Wv,   Wvb, nHH4);
        cvt_bf16_kernel<<<(nHH4 + 255) / 256, blk, 0, stream>>>(Wo,   Wob, nHH4);
        cvt_bf16_kernel<<<(nF4  + 255) / 256, blk, 0, stream>>>(fc1w, F1b, nF4);
        cvt_bf16_kernel<<<(nF4  + 255) / 256, blk, 0, stream>>>(fc2w, F2b, nF4);
    }

    embed_kernel<<<M_, blk, 0, stream>>>(x, emb, z);

    for (int i = 0; i < NL_; ++i) {
        const size_t oHH = (size_t)i * H_ * H_;
        layernorm_kernel<<<M_, blk, 0, stream>>>(z, ln1w + i * H_, ln1b + i * H_, zn);
        gemm_bf16_kernel<false, false, true><<<gH, blk, 0, stream>>>(zn, Wqb + oHH, nullptr, nullptr, nullptr, qbuf, H_, H_);
        gemm_bf16_kernel<false, false, true><<<gH, blk, 0, stream>>>(zn, Wkb + oHH, nullptr, nullptr, nullptr, kbuf, H_, H_);
        gemm_bf16_kernel<false, false, true><<<gH, blk, 0, stream>>>(zn, Wvb + oHH, nullptr, nullptr, nullptr, vbuf, H_, H_);
        attn_kernel<<<gA, dim3(32), 0, stream>>>(qbuf, kbuf, vbuf, length, cbuf);
        gemm_bf16_kernel<false, true, false><<<gH, blk, 0, stream>>>(cbuf, Wob + oHH, nullptr, z, z, nullptr, H_, H_);
        layernorm_kernel<<<M_, blk, 0, stream>>>(z, ln2w + i * H_, ln2b + i * H_, zn);
        gemm_bf16_kernel<true, false, true><<<gF, blk, 0, stream>>>(zn, F1b + (size_t)i * 4 * H_ * H_,
                                                                    fc1b + (size_t)i * 4 * H_,
                                                                    nullptr, nullptr, fbuf, 4 * H_, H_);
        gemm_bf16_kernel<false, true, false><<<gH, blk, 0, stream>>>(fbuf, F2b + (size_t)i * H_ * 4 * H_,
                                                                     fc2b + (size_t)i * H_,
                                                                     z, z, nullptr, H_, 4 * H_);
    }

    final_kernel<<<B_, blk, 0, stream>>>(z, glnw, glnb, gw, gb, out);
}
